// DeepSeek_21174188769507
// MI455X (gfx1250) — matrix operand order repaired and checked by reading, not compiled, not run on hardware
//
#include <hip/hip_runtime.h>
#include <hip/hip_bf16.h>

// ---------------- model dims (fixed by the reference) ----------------
#define S_  1024
#define H_  4096
#define NH_ 32
#define HD_ 128
#define F_  11008
#define L_  2
#define O_  512

typedef __bf16 bf16;
typedef __attribute__((ext_vector_type(16))) __bf16 v16bf;
typedef __attribute__((ext_vector_type(8)))  __bf16 v8bf;
typedef __attribute__((ext_vector_type(8)))  float  v8f;
typedef __attribute__((ext_vector_type(4)))  float  v4f;

// native round-to-nearest-even convert -- lowers to v_cvt_*_bf16_f32 hardware ops
__device__ __forceinline__ bf16 f2bf(float f) { return (bf16)f; }

// ---------------------------------------------------------------------
// one-shot fp32 -> bf16 conversion (weights staged once per GEMM use)
// ---------------------------------------------------------------------
__global__ __launch_bounds__(256)
void f32_to_bf16_kernel(const float* __restrict__ x, bf16* __restrict__ y, int n8) {
  int i = blockIdx.x * blockDim.x + threadIdx.x;
  if (i >= n8) return;
  const size_t base = (size_t)i * 8;
  v4f a = *(const v4f*)(x + base);
  v4f b = *(const v4f*)(x + base + 4);
  v8bf o;
#pragma unroll
  for (int j = 0; j < 4; ++j) { o[j] = f2bf(a[j]); o[4 + j] = f2bf(b[j]); }
  *(v8bf*)(y + base) = o;
}

// ---------------------------------------------------------------------
// GEMM: C[M,N] = A[M,K](bf16 row-major) x B[K,N](bf16 row-major)
// Block = 128 threads = 4 waves; each wave owns a 32x64 C strip
// (2 M-frags x 4 N-frags = 8 accumulators). Waves stack along M.
// Per K-step (32): 2 A-frags (4 x b128) + 4 B-frags (8 x b128) loaded into
// DISTINCT registers first, then 8 x v_wmma_f32_16x16x32_bf16 -- lets the
// scheduler use partial s_wait_loadcnt and overlap WMMA with loads.
// ---------------------------------------------------------------------
__global__ __launch_bounds__(128)
void gemm_bf16_wmma(const bf16* __restrict__ A, const bf16* __restrict__ B,
                    float* __restrict__ C, const float* __restrict__ bias,
                    int M, int N, int K) {
  const int lane = threadIdx.x & 31;
  const int wave = threadIdx.x >> 5;
  const int m0 = blockIdx.y * 128 + wave * 32;  // 32 rows per wave
  const int n0 = blockIdx.x * 64;               // 64 cols per wave
  const int g  = lane >> 4;                     // half-wave group (0/1)
  const int mr = lane & 15;

  v8f acc[8];
#pragma unroll
  for (int t = 0; t < 8; ++t) acc[t] = (v8f){};

  // A fragments: rows m0+mr and m0+16+mr; K chunks g*8+{0..7}, 16+g*8+{0..7}
  const bf16* __restrict__ arow0 = A + (size_t)(m0 + mr) * K + g * 8;
  const bf16* __restrict__ arow1 = arow0 + (size_t)16 * K;
  // B fragment: lane l supplies column N = n0+t*16+l%16, K rows g*8+{0..7}, 16+g*8+{0..7}
  const bf16* __restrict__ brow0 = B + (size_t)(g * 8) * N + n0 + mr;

#pragma unroll 2
  for (int k0 = 0; k0 < K; k0 += 32) {
    const bf16* __restrict__ brow = brow0 + (size_t)k0 * N;
    __builtin_prefetch(brow + (size_t)32 * N, 0, 1);  // next K-step (speculative)

    // ---- issue ALL loads of this K-step into distinct registers ----
    v8bf a0lo = *(const v8bf*)(arow0 + k0);
    v8bf a0hi = *(const v8bf*)(arow0 + k0 + 16);
    v8bf a1lo = *(const v8bf*)(arow1 + k0);
    v8bf a1hi = *(const v8bf*)(arow1 + k0 + 16);
    v16bf bf0, bf1, bf2, bf3;
#pragma unroll
    for (int s = 0; s < 16; ++s) {
      const size_t ko = (size_t)((s < 8) ? s : 8 + s) * N;
      bf0[s] = brow[ko];
      bf1[s] = brow[ko + 16];
      bf2[s] = brow[ko + 32];
      bf3[s] = brow[ko + 48];
    }

    v16bf af0 = __builtin_shufflevector(a0lo, a0hi,
        0, 1, 2, 3, 4, 5, 6, 7, 8, 9, 10, 11, 12, 13, 14, 15);
    v16bf af1 = __builtin_shufflevector(a1lo, a1hi,
        0, 1, 2, 3, 4, 5, 6, 7, 8, 9, 10, 11, 12, 13, 14, 15);


    // ---- 8 WMMAs; each B-frag feeds two M-frags ----
    acc[0] = __builtin_amdgcn_wmma_f32_16x16x32_bf16(
        false, af0, false, bf0, (short)0, acc[0], false, false);
    acc[4] = __builtin_amdgcn_wmma_f32_16x16x32_bf16(
        false, af1, false, bf0, (short)0, acc[4], false, false);
    acc[1] = __builtin_amdgcn_wmma_f32_16x16x32_bf16(
        false, af0, false, bf1, (short)0, acc[1], false, false);
    acc[5] = __builtin_amdgcn_wmma_f32_16x16x32_bf16(
        false, af1, false, bf1, (short)0, acc[5], false, false);
    acc[2] = __builtin_amdgcn_wmma_f32_16x16x32_bf16(
        false, af0, false, bf2, (short)0, acc[2], false, false);
    acc[6] = __builtin_amdgcn_wmma_f32_16x16x32_bf16(
        false, af1, false, bf2, (short)0, acc[6], false, false);
    acc[3] = __builtin_amdgcn_wmma_f32_16x16x32_bf16(
        false, af0, false, bf3, (short)0, acc[3], false, false);
    acc[7] = __builtin_amdgcn_wmma_f32_16x16x32_bf16(
        false, af1, false, bf3, (short)0, acc[7], false, false);
  }

  // C/D layout: lanes 0-15 -> M = mbase+0..7 (vgpr r), lanes 16-31 -> M = mbase+8..15
#pragma unroll
  for (int t = 0; t < 4; ++t) {
    const int n = n0 + t * 16 + mr;
    const float badd = bias ? bias[n] : 0.0f;
    const int mb0 = m0 + g * 8;
    const int mb1 = m0 + 16 + g * 8;
#pragma unroll
    for (int r = 0; r < 8; ++r) {
      C[(size_t)(mb0 + r) * N + n] = acc[t][r] + badd;
      C[(size_t)(mb1 + r) * N + n] = acc[4 + t][r] + badd;
    }
  }
}

// ---------------- embedding gather ----------------
__global__ void embed_kernel(const int* __restrict__ x,
                             const float* __restrict__ E,
                             float* __restrict__ h) {
  int idx = blockIdx.x * blockDim.x + threadIdx.x;
  if (idx >= S_ * H_) return;
  int s = idx / H_, j = idx - s * H_;
  h[idx] = E[(size_t)x[s] * H_ + j];
}

// ---------------- RMSNorm(fp32) -> bf16 ----------------
__global__ __launch_bounds__(256)
void rmsnorm_bf16_kernel(const float* __restrict__ X,
                         const float* __restrict__ w,
                         bf16* __restrict__ Y) {
  __shared__ float red[8];
  const int row = blockIdx.x;
  const float* x = X + (size_t)row * H_;
  float ss = 0.f;
  for (int j = threadIdx.x; j < H_; j += 256) { float v = x[j]; ss += v * v; }
#pragma unroll
  for (int off = 16; off; off >>= 1) ss += __shfl_xor(ss, off, 32);
  if ((threadIdx.x & 31) == 0) red[threadIdx.x >> 5] = ss;
  __syncthreads();
  float tot = 0.f;
#pragma unroll
  for (int i = 0; i < 8; ++i) tot += red[i];
  const float scale = rsqrtf(tot / (float)H_ + 1e-6f);
  bf16* y = Y + (size_t)row * H_;
  for (int j = threadIdx.x; j < H_; j += 256)
    y[j] = f2bf(x[j] * scale * w[j]);
}

// ---------------- RoPE (in place on q and k) ----------------
__global__ void rope_kernel(float* __restrict__ q, float* __restrict__ k) {
  const int half = HD_ / 2;  // 64
  int idx = blockIdx.x * blockDim.x + threadIdx.x;
  if (idx >= S_ * NH_ * half) return;
  int i = idx % half;
  int h = (idx / half) % NH_;
  int s = idx / (half * NH_);
  float inv = expf(-((2.0f * (float)i) / (float)HD_) * logf(10000.0f));
  float ang = (float)s * inv;
  float c = cosf(ang), sn = sinf(ang);
  size_t base = (size_t)s * H_ + (size_t)h * HD_;
  float x1 = q[base + i], x2 = q[base + half + i];
  q[base + i]        = x1 * c - x2 * sn;
  q[base + half + i] = x2 * c + x1 * sn;
  x1 = k[base + i]; x2 = k[base + half + i];
  k[base + i]        = x1 * c - x2 * sn;
  k[base + half + i] = x2 * c + x1 * sn;
}

// ---------------- causal softmax attention, output as bf16 ----------------
// 8 waves per block; each wave handles one (query-row, head) pair.
__global__ __launch_bounds__(256)
void attn_kernel(const float* __restrict__ Q, const float* __restrict__ K,
                 const float* __restrict__ V, bf16* __restrict__ Ob) {
  __shared__ float sc[8][S_];    // 32 KB
  __shared__ float qv[8][HD_];   // 4 KB
  const int lane = threadIdx.x & 31;
  const int w    = threadIdx.x >> 5;
  const int rid  = blockIdx.x * 8 + w;
  const int hh   = rid & (NH_ - 1);
  const int s    = rid >> 5;                 // NH_ == 32
  const size_t qbase = (size_t)s * H_ + (size_t)hh * HD_;

  for (int j = lane; j < HD_; j += 32) qv[w][j] = Q[qbase + j];
  __syncthreads();

  const float scale = 0.0883883476483184f;   // 1/sqrt(128)
  float mx = -1e30f;
  for (int kk = lane; kk <= s; kk += 32) {
    const float* kr = K + (size_t)kk * H_ + (size_t)hh * HD_;
    float d = 0.f;
#pragma unroll 8
    for (int j = 0; j < HD_; ++j) d += qv[w][j] * kr[j];
    d *= scale;
    sc[w][kk] = d;
    mx = fmaxf(mx, d);
  }
#pragma unroll
  for (int off = 16; off; off >>= 1) mx = fmaxf(mx, __shfl_xor(mx, off, 32));

  float sum = 0.f;
  for (int kk = lane; kk <= s; kk += 32) {
    float p = expf(sc[w][kk] - mx);
    sc[w][kk] = p;
    sum += p;
  }
#pragma unroll
  for (int off = 16; off; off >>= 1) sum += __shfl_xor(sum, off, 32);
  const float rs = 1.0f / sum;
  __syncthreads();

  float acc[4] = {0.f, 0.f, 0.f, 0.f};
  for (int kk = 0; kk <= s; ++kk) {
    const float p = sc[w][kk];
    const float* vr = V + (size_t)kk * H_ + (size_t)hh * HD_;
#pragma unroll
    for (int j = 0; j < 4; ++j) acc[j] += p * vr[lane + 32 * j];
  }
#pragma unroll
  for (int j = 0; j < 4; ++j)
    Ob[qbase + lane + 32 * j] = f2bf(acc[j] * rs);
}

// ---------------- elementwise ----------------
__global__ void add_inplace_kernel(float* __restrict__ h,
                                   const float* __restrict__ o, int n) {
  int i = blockIdx.x * blockDim.x + threadIdx.x;
  if (i < n) h[i] += o[i];
}

__global__ void silu_mul_bf16_kernel(const float* __restrict__ g,
                                     const float* __restrict__ u,
                                     bf16* __restrict__ y, int n) {
  int i = blockIdx.x * blockDim.x + threadIdx.x;
  if (i < n) {
    float gv = g[i];
    float sv = gv / (1.0f + expf(-gv));
    y[i] = f2bf(sv * u[i]);
  }
}

// ---------------------------------------------------------------------
extern "C" void kernel_launch(void* const* d_in, const int* in_sizes, int n_in,
                              void* d_out, int out_size, void* d_ws, size_t ws_size,
                              hipStream_t stream) {
  (void)in_sizes; (void)n_in; (void)out_size; (void)ws_size;

  const int*   x     = (const int*)d_in[0];
  const float* embed = (const float*)d_in[1];
  const float* Wq    = (const float*)d_in[2];
  const float* Wk    = (const float*)d_in[3];
  const float* Wv    = (const float*)d_in[4];
  const float* Wo    = (const float*)d_in[5];
  const float* Wg    = (const float*)d_in[6];
  const float* Wu    = (const float*)d_in[7];
  const float* Wd    = (const float*)d_in[8];
  const float* ln1   = (const float*)d_in[9];
  const float* ln2   = (const float*)d_in[10];
  const float* lnf   = (const float*)d_in[11];
  const float* Wout  = (const float*)d_in[12];
  const float* bout  = (const float*)d_in[13];
  float* out = (float*)d_out;

  // carve workspace (~300 MB total)
  char* wp = (char*)d_ws;
  auto carve = [&](size_t bytes) -> void* {
    void* p = (void*)wp;
    wp += (bytes + 255) & ~(size_t)255;
    return p;
  };
  float* h      = (float*)carve(sizeof(float) * S_ * H_);
  float* q      = (float*)carve(sizeof(float) * S_ * H_);
  float* k      = (float*)carve(sizeof(float) * S_ * H_);
  float* v      = (float*)carve(sizeof(float) * S_ * H_);
  float* o      = (float*)carve(sizeof(float) * S_ * H_);   // attn-proj / FFN-down out
  float* gbuf   = (float*)carve(sizeof(float) * S_ * F_);
  float* ubuf   = (float*)carve(sizeof(float) * S_ * F_);
  bf16*  a_bf   = (bf16*)carve(sizeof(bf16) * S_ * H_);     // rmsnorm outputs
  bf16*  att_bf = (bf16*)carve(sizeof(bf16) * S_ * H_);     // attention output
  bf16*  act_bf = (bf16*)carve(sizeof(bf16) * S_ * F_);     // silu(g)*u
  bf16*  wstage = (bf16*)carve(sizeof(bf16) * (size_t)H_ * F_);  // bf16 weight staging (90 MB)

  // stage weight -> bf16 once, then WMMA GEMM against it
  auto gemm = [&](const bf16* A, const float* W, float* C, const float* bias,
                  int M, int N, int K) {
    const int n8 = (K * N) / 8;
    f32_to_bf16_kernel<<<(n8 + 255) / 256, 256, 0, stream>>>(W, wstage, n8);
    dim3 grid(N / 64, M / 128);
    gemm_bf16_wmma<<<grid, dim3(128), 0, stream>>>(A, wstage, C, bias, M, N, K);
  };

  // embedding
  embed_kernel<<<(S_ * H_) / 256, 256, 0, stream>>>(x, embed, h);

  for (int l = 0; l < L_; ++l) {
    const float* wq = Wq + (size_t)l * H_ * H_;
    const float* wk = Wk + (size_t)l * H_ * H_;
    const float* wv = Wv + (size_t)l * H_ * H_;
    const float* wo = Wo + (size_t)l * H_ * H_;
    const float* wg = Wg + (size_t)l * H_ * F_;
    const float* wu = Wu + (size_t)l * H_ * F_;
    const float* wd = Wd + (size_t)l * F_ * H_;

    // attention block
    rmsnorm_bf16_kernel<<<S_, 256, 0, stream>>>(h, ln1 + (size_t)l * H_, a_bf);
    gemm(a_bf, wq, q, nullptr, S_, H_, H_);
    gemm(a_bf, wk, k, nullptr, S_, H_, H_);
    gemm(a_bf, wv, v, nullptr, S_, H_, H_);
    rope_kernel<<<(S_ * NH_ * (HD_ / 2)) / 256, 256, 0, stream>>>(q, k);
    attn_kernel<<<(S_ * NH_) / 8, 256, 0, stream>>>(q, k, v, att_bf);
    gemm(att_bf, wo, o, nullptr, S_, H_, H_);
    add_inplace_kernel<<<(S_ * H_) / 256, 256, 0, stream>>>(h, o, S_ * H_);

    // SwiGLU FFN
    rmsnorm_bf16_kernel<<<S_, 256, 0, stream>>>(h, ln2 + (size_t)l * H_, a_bf);
    gemm(a_bf, wg, gbuf, nullptr, S_, F_, H_);
    gemm(a_bf, wu, ubuf, nullptr, S_, F_, H_);
    silu_mul_bf16_kernel<<<(S_ * F_ + 255) / 256, 256, 0, stream>>>(gbuf, ubuf,
                                                                    act_bf, S_ * F_);
    gemm(act_bf, wd, o, nullptr, S_, H_, F_);
    add_inplace_kernel<<<(S_ * H_) / 256, 256, 0, stream>>>(h, o, S_ * H_);
  }

  // final norm + output head (writes d_out, with bias)
  rmsnorm_bf16_kernel<<<S_, 256, 0, stream>>>(h, lnf, a_bf);
  gemm(a_bf, Wout, out, bout, S_, O_, H_);
}
